// DeepEC_KAN_77103252898334
// MI455X (gfx1250) — matrix operand order repaired and checked by reading, not compiled, not run on hardware
//
#include <hip/hip_runtime.h>
#include <hip/hip_bf16.h>

typedef _Float16 h8   __attribute__((ext_vector_type(8)));
typedef _Float16 v16h __attribute__((ext_vector_type(16)));
typedef float    v8f  __attribute__((ext_vector_type(8)));

// Build a v16h A/B fragment from two 16-byte aligned 8-half chunks.
static __device__ __forceinline__ v16h ld_frag(const _Float16* p0, const _Float16* p1) {
    h8 lo = *(const h8*)p0;
    h8 hi = *(const h8*)p1;
    return __builtin_shufflevector(lo, hi, 0,1,2,3,4,5,6,7,8,9,10,11,12,13,14,15);
}

static __device__ __forceinline__ v8f wmma_f16(v16h a, v16h b, v8f c) {
    return __builtin_amdgcn_wmma_f32_16x16x32_f16(false, a, false, b, (short)0, c, false, false);
}

// ---------------------------------------------------------------------------
// Kernel 1: conv1d (VALID) + bias + ReLU + global max-pool + BatchNorm.
// One block per batch element. 8 waves; wave w owns out-channels [16w,16w+16).
// x is staged to LDS as f16 with channels padded 21 -> 24 so that the im2col
// matrix B[r,l] = xpad[l*24 + r] gives 16B-aligned ds_load_b128 fragments and
// reduction length RP = 24*KTAPS is a multiple of 32.
// A (weight) fragments are hoisted into registers once -> 2 ds_load_b128 per
// WMMA in the steady-state loop (B only).
// ---------------------------------------------------------------------------
template<int KTAPS>
__global__ void __launch_bounds__(256) conv_relu_max_bn(
    const float* __restrict__ x,     // [256,1000,21]
    const float* __restrict__ w,     // [128,21,KTAPS]
    const float* __restrict__ bias,  // [128]
    const float* __restrict__ bng, const float* __restrict__ bnb,
    const float* __restrict__ bnm, const float* __restrict__ bnv,
    float* __restrict__ h, int h_off)
{
    constexpr int RP   = 24 * KTAPS;          // padded reduction (96/192/384)
    constexpr int KI   = RP / 32;             // WMMA K-steps (3/6/12)
    constexpr int LOUT = 1000 - KTAPS + 1;
    constexpr int NT   = (LOUT + 15) / 16;
    extern __shared__ char smem[];
    _Float16* xp = (_Float16*)smem;                     // [1024*24] padded x
    _Float16* wa = (_Float16*)(smem + 1024 * 24 * 2);   // [128*RP]   padded A

    const int tid = threadIdx.x;
    const int b   = blockIdx.x;

    for (int i = tid; i < 1024 * 24; i += 256) xp[i] = (_Float16)0.0f;
    for (int i = tid; i < 128 * RP;  i += 256) wa[i] = (_Float16)0.0f;
    __syncthreads();

    const float* xb = x + (size_t)b * 21000;
    for (int i = tid; i < 1000 * 21; i += 256) {
        int l = i / 21, c = i % 21;
        xp[l * 24 + c] = (_Float16)xb[i];
    }
    for (int i = tid; i < 128 * 21 * KTAPS; i += 256) {
        int o = i / (21 * KTAPS);
        int r = i % (21 * KTAPS);
        int c = r / KTAPS, k = r % KTAPS;
        wa[o * RP + k * 24 + c] = (_Float16)w[i];   // r' = k*24 + c matches xpad
    }
    __syncthreads();

    const int wave = tid >> 5;
    const int lane = tid & 31;
    const int lh   = lane >> 4;     // lane half (0/1)
    const int ll   = lane & 15;

    float bvreg[8];
#pragma unroll
    for (int v = 0; v < 8; ++v) bvreg[v] = bias[wave * 16 + v + lh * 8];

    float mx[8];
#pragma unroll
    for (int v = 0; v < 8; ++v) mx[v] = 0.0f;   // ReLU output >= 0

    // Hoist all A fragments for this wave's 16-row block into registers.
    const _Float16* arow = wa + (size_t)(wave * 16 + ll) * RP;
    v16h afr[KI];
#pragma unroll
    for (int ki = 0; ki < KI; ++ki) {
        const _Float16* ap = arow + ki * 32 + lh * 8;   // lo={0..7,16..23}, hi={8..15,24..31}
        afr[ki] = ld_frag(ap, ap + 16);
    }

    for (int t = 0; t < NT; ++t) {
        const int l0 = t * 16;
        const _Float16* bcol = xp + (size_t)(l0 + ll) * 24 + lh * 8;
        v8f acc = {};
#pragma unroll
        for (int ki = 0; ki < KI; ++ki) {
            const _Float16* bp = bcol + ki * 32;        // B: col-per-lane, lo={0..7,16..23}, hi={8..15,24..31}
            v16h bf = ld_frag(bp, bp + 16);
            acc = wmma_f16(afr[ki], bf, acc);
        }
        const bool valid = (l0 + ll) < LOUT;
#pragma unroll
        for (int v = 0; v < 8; ++v) {
            float val = acc[v] + bvreg[v];
            val = val > 0.0f ? val : 0.0f;
            if (valid && val > mx[v]) mx[v] = val;
        }
    }

#pragma unroll
    for (int v = 0; v < 8; ++v) {
        float m = mx[v];
        m = fmaxf(m, __shfl_xor(m, 1, 32));
        m = fmaxf(m, __shfl_xor(m, 2, 32));
        m = fmaxf(m, __shfl_xor(m, 4, 32));
        m = fmaxf(m, __shfl_xor(m, 8, 32));
        if (ll == 0) {
            int M = wave * 16 + v + lh * 8;
            float o = (m - bnm[M]) * rsqrtf(bnv[M] + 1e-5f) * bng[M] + bnb[M];
            h[(size_t)b * 384 + h_off + M] = o;
        }
    }
}

// ---------------------------------------------------------------------------
// Cubic B-spline bases, grid_size=3, k=3 -> 6 bases. Uniform grid, step 2/3.
// ---------------------------------------------------------------------------
static __device__ __forceinline__ void bspline6(float xv, float* out) {
    const float H = 2.0f / 3.0f;
    float gr[10];
#pragma unroll
    for (int j = 0; j < 10; ++j) gr[j] = -3.0f + H * (float)j;
    float b0[9];
#pragma unroll
    for (int j = 0; j < 9; ++j) b0[j] = (xv >= gr[j] && xv < gr[j + 1]) ? 1.0f : 0.0f;
    float b1[8];
#pragma unroll
    for (int j = 0; j < 8; ++j)
        b1[j] = (xv - gr[j]) / (1.0f * H) * b0[j] + (gr[j + 2] - xv) / (1.0f * H) * b0[j + 1];
    float b2[7];
#pragma unroll
    for (int j = 0; j < 7; ++j)
        b2[j] = (xv - gr[j]) / (2.0f * H) * b1[j] + (gr[j + 3] - xv) / (2.0f * H) * b1[j + 1];
#pragma unroll
    for (int j = 0; j < 6; ++j)
        out[j] = (xv - gr[j]) / (3.0f * H) * b2[j] + (gr[j + 4] - xv) / (3.0f * H) * b2[j + 1];
}

// ---------------------------------------------------------------------------
// Kernel 2a: BN4 + LayerNorm + KAN feature expansion -> E1[b][2688] f16
//            E1 = [ silu(xn)[384] | bases(xn)[384*6] ]
// One block per batch row, 384 threads (12 waves).
// ---------------------------------------------------------------------------
__global__ void __launch_bounds__(384) expand_kan1(
    const float* __restrict__ h,
    const float* __restrict__ g4, const float* __restrict__ b4,
    const float* __restrict__ m4, const float* __restrict__ v4,
    const float* __restrict__ lng, const float* __restrict__ lnb,
    _Float16* __restrict__ E)
{
    __shared__ float red[32];
    const int b = blockIdx.x, i = threadIdx.x;
    const int wid = i >> 5, lane = i & 31;

    float val = h[(size_t)b * 384 + i];
    val = (val - m4[i]) * rsqrtf(v4[i] + 1e-5f) * g4[i] + b4[i];

    float s = val;
    s += __shfl_xor(s, 16, 32); s += __shfl_xor(s, 8, 32);
    s += __shfl_xor(s, 4, 32);  s += __shfl_xor(s, 2, 32); s += __shfl_xor(s, 1, 32);
    if (lane == 0) red[wid] = s;
    __syncthreads();
    if (i == 0) { float t = 0.f; for (int k2 = 0; k2 < 12; ++k2) t += red[k2]; red[16] = t * (1.0f / 384.0f); }
    __syncthreads();
    const float mu = red[16];
    const float d  = val - mu;
    __syncthreads();

    float s2 = d * d;
    s2 += __shfl_xor(s2, 16, 32); s2 += __shfl_xor(s2, 8, 32);
    s2 += __shfl_xor(s2, 4, 32);  s2 += __shfl_xor(s2, 2, 32); s2 += __shfl_xor(s2, 1, 32);
    if (lane == 0) red[wid] = s2;
    __syncthreads();
    if (i == 0) { float t = 0.f; for (int k2 = 0; k2 < 12; ++k2) t += red[k2]; red[17] = t * (1.0f / 384.0f); }
    __syncthreads();
    const float var = red[17];

    const float xn = d * rsqrtf(var + 1e-5f) * lng[i] + lnb[i];

    _Float16* Er = E + (size_t)b * 2688;
    const float sil = xn / (1.0f + __expf(-xn));
    Er[i] = (_Float16)sil;
    float bs[6];
    bspline6(xn, bs);
#pragma unroll
    for (int g = 0; g < 6; ++g) Er[384 + i * 6 + g] = (_Float16)bs[g];
}

// Kernel 2c: KAN2 feature expansion (no norm) -> E2[b][3584]
__global__ void __launch_bounds__(512) expand_kan2(
    const float* __restrict__ h2, _Float16* __restrict__ E)
{
    const int b = blockIdx.x, i = threadIdx.x;
    const float xv = h2[(size_t)b * 512 + i];
    _Float16* Er = E + (size_t)b * 3584;
    Er[i] = (_Float16)(xv / (1.0f + __expf(-xv)));
    float bs[6];
    bspline6(xv, bs);
#pragma unroll
    for (int g = 0; g < 6; ++g) Er[512 + i * 6 + g] = (_Float16)bs[g];
}

// ---------------------------------------------------------------------------
// Weight packers: fuse [base | spline] weights into f16, K-contiguous per
// output feature (GEMM "B" rows).
// ---------------------------------------------------------------------------
__global__ void pack_w1(const float* __restrict__ bw, const float* __restrict__ sw,
                        _Float16* __restrict__ W)
{
    int idx = blockIdx.x * 256 + threadIdx.x;
    if (idx >= 512 * 2688) return;
    int o = idx / 2688, r = idx % 2688;
    float v = (r < 384) ? bw[o * 384 + r] : sw[o * 2304 + (r - 384)];
    W[idx] = (_Float16)v;
}

__global__ void pack_w2(const float* __restrict__ bw, const float* __restrict__ sw,
                        _Float16* __restrict__ W)
{
    int idx = blockIdx.x * 256 + threadIdx.x;
    if (idx >= 240 * 3584) return;
    int o = idx / 3584, r = idx % 3584;
    float v = 0.0f;
    if (o < 229) v = (r < 512) ? bw[o * 512 + r] : sw[o * 3072 + (r - 512)];
    W[idx] = (_Float16)v;
}

// ---------------------------------------------------------------------------
// Generic WMMA GEMM: C[256][Nout] = A[256][K] * B[Npad][K]^T (f16 in, f32 out)
// Operands stream directly from global memory (entire working set is
// L2-resident: 192 MB L2 >> 8 MB). 8 waves/block, one 16x16 tile per wave.
// ---------------------------------------------------------------------------
__global__ void __launch_bounds__(256) wmma_gemm(
    const _Float16* __restrict__ A, const _Float16* __restrict__ B,
    float* __restrict__ C, int K, int Nout, int Ntiles)
{
    const int tid  = threadIdx.x;
    const int wave = tid >> 5, lane = tid & 31;
    const int lh   = lane >> 4, ll = lane & 15;
    const int tile = blockIdx.x * 8 + wave;
    const int m0   = (tile / Ntiles) * 16;
    const int n0   = (tile % Ntiles) * 16;

    const _Float16* arow = A + (size_t)(m0 + ll) * K;
    const _Float16* brow = B + (size_t)(n0 + ll) * K;

    v8f acc = {};
    for (int k0 = 0; k0 < K; k0 += 32) {
        const _Float16* ap = arow + k0 + lh * 8;
        v16h af = ld_frag(ap, ap + 16);
        const _Float16* bp = brow + k0 + lh * 8;
        v16h bf = ld_frag(bp, bp + 16);
        acc = wmma_f16(af, bf, acc);
    }

    const int n = n0 + ll;
    if (n < Nout) {
#pragma unroll
        for (int v = 0; v < 8; ++v) {
            int m = m0 + v + lh * 8;
            C[(size_t)m * Nout + n] = acc[v];
        }
    }
}

// ---------------------------------------------------------------------------
extern "C" void kernel_launch(void* const* d_in, const int* in_sizes, int n_in,
                              void* d_out, int out_size, void* d_ws, size_t ws_size,
                              hipStream_t stream) {
    (void)in_sizes; (void)n_in; (void)out_size; (void)ws_size;
    const float* x        = (const float*)d_in[0];
    const float* conv1_w  = (const float*)d_in[1];
    const float* conv1_b  = (const float*)d_in[2];
    const float* conv2_w  = (const float*)d_in[3];
    const float* conv2_b  = (const float*)d_in[4];
    const float* conv3_w  = (const float*)d_in[5];
    const float* conv3_b  = (const float*)d_in[6];
    const float* bn1_g = (const float*)d_in[7];  const float* bn1_b = (const float*)d_in[8];
    const float* bn1_m = (const float*)d_in[9];  const float* bn1_v = (const float*)d_in[10];
    const float* bn2_g = (const float*)d_in[11]; const float* bn2_b = (const float*)d_in[12];
    const float* bn2_m = (const float*)d_in[13]; const float* bn2_v = (const float*)d_in[14];
    const float* bn3_g = (const float*)d_in[15]; const float* bn3_b = (const float*)d_in[16];
    const float* bn3_m = (const float*)d_in[17]; const float* bn3_v = (const float*)d_in[18];
    const float* bn4_g = (const float*)d_in[19]; const float* bn4_b = (const float*)d_in[20];
    const float* bn4_m = (const float*)d_in[21]; const float* bn4_v = (const float*)d_in[22];
    const float* ln_g  = (const float*)d_in[23]; const float* ln_b  = (const float*)d_in[24];
    const float* base_w1   = (const float*)d_in[25];
    const float* spline_w1 = (const float*)d_in[26];
    const float* base_w2   = (const float*)d_in[27];
    const float* spline_w2 = (const float*)d_in[28];

    char* ws = (char*)d_ws;
    float*    h   = (float*)   (ws + 0);         // 256*384*4   = 393216
    _Float16* E1  = (_Float16*)(ws + 393216);    // 256*2688*2  = 1376256
    _Float16* W1  = (_Float16*)(ws + 1769472);   // 512*2688*2  = 2752512
    float*    h2  = (float*)   (ws + 4521984);   // 256*512*4   = 524288
    _Float16* E2  = (_Float16*)(ws + 5046272);   // 256*3584*2  = 1835008
    _Float16* W2  = (_Float16*)(ws + 6881280);   // 240*3584*2  = 1720320

    // Conv + ReLU + maxpool + BN -> h[256,384]
    conv_relu_max_bn<4><<<256, 256, 1024*24*2 + 128*96*2,  stream>>>(
        x, conv1_w, conv1_b, bn1_g, bn1_b, bn1_m, bn1_v, h, 0);
    conv_relu_max_bn<8><<<256, 256, 1024*24*2 + 128*192*2, stream>>>(
        x, conv2_w, conv2_b, bn2_g, bn2_b, bn2_m, bn2_v, h, 128);
    conv_relu_max_bn<16><<<256, 256, 1024*24*2 + 128*384*2, stream>>>(
        x, conv3_w, conv3_b, bn3_g, bn3_b, bn3_m, bn3_v, h, 256);

    // Pack KAN weights to f16 (K-contiguous rows)
    pack_w1<<<(512 * 2688 + 255) / 256, 256, 0, stream>>>(base_w1, spline_w1, W1);
    pack_w2<<<(240 * 3584 + 255) / 256, 256, 0, stream>>>(base_w2, spline_w2, W2);

    // BN4 + LN + KAN1 expansion, then GEMM [256,2688]x[2688,512]
    expand_kan1<<<256, 384, 0, stream>>>(h, bn4_g, bn4_b, bn4_m, bn4_v, ln_g, ln_b, E1);
    wmma_gemm<<<64, 256, 0, stream>>>(E1, W1, h2, 2688, 512, 32);

    // KAN2 expansion, then GEMM [256,3584]x[3584,229] -> output
    expand_kan2<<<256, 512, 0, stream>>>(h2, E2);
    wmma_gemm<<<30, 256, 0, stream>>>(E2, W2, (float*)d_out, 3584, 229, 15);
}
